// NonLocalBlockLinear_46162308497531
// MI455X (gfx1250) — matrix operand order repaired and checked by reading, not compiled, not run on hardware
//
#include <hip/hip_runtime.h>

// ---------------------------------------------------------------------------
// NonLocalBlock (linear-embedded non-local attention), MI455X / gfx1250.
// All matrix math in bf16 WMMA (v_wmma_f32_16x16x32_bf16, f32 accumulate).
// Flash-attention online softmax; K/V tiles staged into LDS per block via
// CDNA5 async global->LDS copies (ASYNCcnt), double-buffered.
// ---------------------------------------------------------------------------

typedef __attribute__((ext_vector_type(16))) __bf16 v16bf;
typedef __attribute__((ext_vector_type(8)))  __bf16 v8bf;
typedef __attribute__((ext_vector_type(8)))  float  v8f;

#define NB   4      // batch
#define CIN  256    // in channels
#define CI   128    // inter channels
#define HW   4096   // N = H*W
#define LDA  264    // LDS row stride (bf16 elems), 16B aligned, bank-skewed

static constexpr float EPS = 1e-5f;
// softmax scale 1/sqrt(CI), folded with log2(e) so we can use exp2
static constexpr float SCALE_L2E = 0.08838834764831845f * 1.4426950408889634f;

__device__ __forceinline__ v16bf cat8(v8bf lo, v8bf hi) {
  return __builtin_shufflevector(lo, hi, 0,1,2,3,4,5,6,7,8,9,10,11,12,13,14,15);
}

__device__ __forceinline__ v8f wmma_bf16(v16bf a, v16bf b, v8f c) {
  return __builtin_amdgcn_wmma_f32_16x16x32_bf16(false, a, false, b, (short)0, c, false, false);
}

// Async global -> LDS copy, 16 bytes per lane (GVS mode: SGPR64 base +
// per-lane VGPR32 byte offset; per-lane LDS byte offset). Tracked by ASYNCcnt.
__device__ __forceinline__ void async_ld16(const void* gbase, unsigned goff,
                                           unsigned ldsoff) {
  asm volatile("global_load_async_to_lds_b128 %0, %1, %2"
               :: "v"(ldsoff), "v"(goff), "s"(gbase) : "memory");
}
__device__ __forceinline__ void wait_async0() {
  asm volatile("s_wait_asynccnt 0x0" ::: "memory");
}

// ---------------------------------------------------------------------------
// Kernel 0: fold BN1d into theta/phi weights, convert weights to bf16,
// precompute biases and output-BN affine terms.
// ---------------------------------------------------------------------------
__global__ void prep_kernel(
    const float* __restrict__ theta_w, const float* __restrict__ theta_b,
    const float* __restrict__ t_g, const float* __restrict__ t_be,
    const float* __restrict__ t_m, const float* __restrict__ t_v,
    const float* __restrict__ phi_w, const float* __restrict__ phi_b,
    const float* __restrict__ p_g, const float* __restrict__ p_be,
    const float* __restrict__ p_m, const float* __restrict__ p_v,
    const float* __restrict__ g_w, const float* __restrict__ g_b,
    const float* __restrict__ wz_w, const float* __restrict__ wz_b,
    const float* __restrict__ bn_g, const float* __restrict__ bn_be,
    const float* __restrict__ bn_m, const float* __restrict__ bn_v,
    __bf16* __restrict__ Wq, __bf16* __restrict__ Wk,
    __bf16* __restrict__ Wv, __bf16* __restrict__ Wz,
    float* __restrict__ bq, float* __restrict__ bk, float* __restrict__ bv,
    float* __restrict__ g1, float* __restrict__ g2) {
  int tid = blockIdx.x * blockDim.x + threadIdx.x;
  int nt  = gridDim.x * blockDim.x;
  for (int i = tid; i < CI * CIN; i += nt) {
    int d = i / CIN;
    float aq = rsqrtf(t_v[d] + EPS) * t_g[d];
    float ak = rsqrtf(p_v[d] + EPS) * p_g[d];
    Wq[i] = (__bf16)(theta_w[i] * aq);
    Wk[i] = (__bf16)(phi_w[i] * ak);
    Wv[i] = (__bf16)(g_w[i]);
    Wz[i] = (__bf16)(wz_w[i]);   // [C, CI] -> same element count
  }
  for (int d = tid; d < CI; d += nt) {
    float aq = rsqrtf(t_v[d] + EPS) * t_g[d];
    float ak = rsqrtf(p_v[d] + EPS) * p_g[d];
    bq[d] = (theta_b[d] - t_m[d]) * aq + t_be[d];
    bk[d] = (phi_b[d]  - p_m[d]) * ak + p_be[d];
    bv[d] = g_b[d];
  }
  for (int c = tid; c < CIN; c += nt) {
    float a = rsqrtf(bn_v[c] + EPS) * bn_g[c];
    g1[c] = a;
    g2[c] = (wz_b[c] - bn_m[c]) * a + bn_be[c];
  }
}

// ---------------------------------------------------------------------------
// Kernel 1: fused Q/K/V projection. Block = 4 waves = 64 spatial rows.
// X^T tile (64 x 256) staged in LDS as bf16; each wave computes 16 rows
// against all 128 output channels for the three projections.
// Q,K stored [B,N,CI] bf16 row-major; V stored transposed [B,CI,N] bf16.
// ---------------------------------------------------------------------------
__global__ __launch_bounds__(128) void qkv_kernel(
    const float* __restrict__ x,
    const __bf16* __restrict__ Wq, const __bf16* __restrict__ Wk,
    const __bf16* __restrict__ Wv,
    const float* __restrict__ bq, const float* __restrict__ bk,
    const float* __restrict__ bv,
    __bf16* __restrict__ Q, __bf16* __restrict__ K,
    __bf16* __restrict__ Vt) {
  __shared__ alignas(16) __bf16 Xs[64 * LDA];
  const int b  = blockIdx.y;
  const int n0 = blockIdx.x * 64;
  const int tid = threadIdx.x;

  // load + transpose x[b, c, n0+r] -> Xs[r][c] (bf16)
  for (int i = tid; i < 64 * CIN; i += 128) {
    int r = i & 63, c = i >> 6;
    Xs[r * LDA + c] = (__bf16)x[((size_t)b * CIN + c) * HW + n0 + r];
  }
  __syncthreads();

  const int lane = tid & 31, wid = tid >> 5;
  const int lm = lane & 15, hl = lane >> 4;
  const int row = wid * 16 + lm;

  // A fragments for this wave's 16 rows, 8 k-chunks of 32 (C=256)
  v16bf af[8];
#pragma unroll
  for (int kc = 0; kc < 8; ++kc) {
    const v8bf* p0 = (const v8bf*)&Xs[row * LDA + kc * 32 + hl * 16];
    const v8bf* p1 = (const v8bf*)&Xs[row * LDA + kc * 32 + hl * 16 + 8];
    af[kc] = cat8(*p0, *p1);
  }

  for (int o = 0; o < 3; ++o) {
    const __bf16* W   = (o == 0) ? Wq : (o == 1) ? Wk : Wv;
    const float* bias = (o == 0) ? bq : (o == 1) ? bk : bv;
#pragma unroll
    for (int t = 0; t < 8; ++t) {
      const int d = t * 16 + lm;          // output channel (B-matrix column)
      v8f acc = {};
#pragma unroll
      for (int kc = 0; kc < 8; ++kc) {
        v16bf bf = *(const v16bf*)&W[(size_t)d * CIN + kc * 32 + hl * 16];
        acc = wmma_bf16(af[kc], bf, acc);
      }
      const float bs = bias[d];
      if (o < 2) {
        __bf16* O = (o == 0) ? Q : K;
#pragma unroll
        for (int v = 0; v < 8; ++v) {
          int n = n0 + wid * 16 + v + hl * 8;
          O[((size_t)b * HW + n) * CI + d] = (__bf16)(acc[v] + bs);
        }
      } else {
#pragma unroll
        for (int v = 0; v < 8; ++v) {
          int n = n0 + wid * 16 + v + hl * 8;
          Vt[((size_t)b * CI + d) * HW + n] = (__bf16)(acc[v] + bs);
        }
      }
    }
  }
}

// ---------------------------------------------------------------------------
// Kernel 2: flash attention. Block = 4 waves; each wave owns 16 query rows;
// key tiles of 32 streamed through double-buffered LDS staging filled with
// async global->LDS copies (shared by all 4 waves -> 4x less L2 traffic).
// S = Q K^T via WMMA (CI=128 -> 4 chunks), online softmax (row stats per
// 16-lane half), P transposed to A-fragment layout via per-wave LDS tile,
// then y += P V via WMMA against the V^T staging tile.
// ---------------------------------------------------------------------------
__global__ __launch_bounds__(128) void attn_kernel(
    const __bf16* __restrict__ Q, const __bf16* __restrict__ K,
    const __bf16* __restrict__ Vt, __bf16* __restrict__ Y) {
  __shared__ alignas(16) __bf16 Ks[2][32 * 128];  // 2 x 8KB  (keys x CI)
  __shared__ alignas(16) __bf16 Vs[2][128 * 32];  // 2 x 8KB  (CI x keys)
  __shared__ alignas(16) __bf16 Ps[4][16 * 32];   // per-wave P transpose tile

  const int b = blockIdx.y;
  const int tid = threadIdx.x;
  const int lane = tid & 31, wid = tid >> 5;
  const int lm = lane & 15, hl = lane >> 4;
  const int qr = blockIdx.x * 64 + wid * 16;    // query row base

  const __bf16* Qb = Q  + (size_t)b * HW * CI;
  const __bf16* Kb = K  + (size_t)b * HW * CI;
  const __bf16* Vb = Vt + (size_t)b * CI * HW;
  __bf16* P = Ps[wid];

  // LDS byte offsets of the staging buffers (low 32 bits of shared pointer)
  const unsigned ksOff[2] = {(unsigned)(size_t)(void*)&Ks[0][0],
                             (unsigned)(size_t)(void*)&Ks[1][0]};
  const unsigned vsOff[2] = {(unsigned)(size_t)(void*)&Vs[0][0],
                             (unsigned)(size_t)(void*)&Vs[1][0]};

  // stage key-tile jt (32 keys) into buffer bi: K tile is 8KB contiguous,
  // V^T tile is 128 rows x 64B (one row per thread).
  auto stage = [&](int j, int bi) {
    unsigned kg = (unsigned)((size_t)j * CI * 2u) + (unsigned)tid * 16u;
#pragma unroll
    for (int i = 0; i < 4; ++i)
      async_ld16(Kb, kg + i * 2048u, ksOff[bi] + (unsigned)tid * 16u + i * 2048u);
    unsigned vg = (unsigned)(((size_t)tid * HW + j) * 2u);
#pragma unroll
    for (int c = 0; c < 4; ++c)
      async_ld16(Vb, vg + c * 16u, vsOff[bi] + (unsigned)tid * 64u + c * 16u);
  };

  // Q A-fragments (persist in registers): 4 k-chunks of 32
  v16bf qf[4];
  {
    const size_t rbase = (size_t)(qr + lm) * CI;
#pragma unroll
    for (int kc = 0; kc < 4; ++kc) {
      const v8bf* p0 = (const v8bf*)&Qb[rbase + kc * 32 + hl * 16];
      const v8bf* p1 = (const v8bf*)&Qb[rbase + kc * 32 + hl * 16 + 8];
      qf[kc] = cat8(*p0, *p1);
    }
  }

  v8f yacc[8];
#pragma unroll
  for (int t = 0; t < 8; ++t) yacc[t] = (v8f){};
  float mrow[8], lrow[8];
#pragma unroll
  for (int v = 0; v < 8; ++v) { mrow[v] = -3.0e38f; lrow[v] = 0.f; }

  stage(0, 0);                               // prime buffer 0

  for (int jt = 0; jt < HW / 32; ++jt) {
    const int bi = jt & 1;
    wait_async0();                           // this wave's staging done
    __syncthreads();                         // whole tile visible block-wide
    if (jt + 1 < HW / 32) stage((jt + 1) * 32, bi ^ 1);

    // ---- S tiles: 16 queries x 32 keys (B-frags from LDS staging) ----
    v8f s0 = {}, s1 = {};
#pragma unroll
    for (int kc = 0; kc < 4; ++kc) {
      v16bf k0 = *(const v16bf*)&Ks[bi][lm * 128 + kc * 32 + hl * 16];
      s0 = wmma_bf16(qf[kc], k0, s0);
    }
#pragma unroll
    for (int kc = 0; kc < 4; ++kc) {
      v16bf k1 = *(const v16bf*)&Ks[bi][(16 + lm) * 128 + kc * 32 + hl * 16];
      s1 = wmma_bf16(qf[kc], k1, s1);
    }

    // ---- online softmax over the 32-key tile -------------------------
#pragma unroll
    for (int v = 0; v < 8; ++v) {
      float a  = s0[v] * SCALE_L2E;
      float bb = s1[v] * SCALE_L2E;
      float tm = fmaxf(a, bb);
#pragma unroll
      for (int off = 8; off; off >>= 1) tm = fmaxf(tm, __shfl_xor(tm, off, 16));
      float mnew = fmaxf(mrow[v], tm);
      float corr = exp2f(mrow[v] - mnew);
      float p0 = exp2f(a - mnew);
      float p1 = exp2f(bb - mnew);
      float rs = p0 + p1;
#pragma unroll
      for (int off = 8; off; off >>= 1) rs += __shfl_xor(rs, off, 16);
      lrow[v] = lrow[v] * corr + rs;
      mrow[v] = mnew;
#pragma unroll
      for (int t = 0; t < 8; ++t) yacc[t][v] *= corr;
      // stash P (bf16) for layout transpose; row = v + hl*8, col = key
      const int r = v + hl * 8;
      P[r * 32 + lm]      = (__bf16)p0;
      P[r * 32 + lm + 16] = (__bf16)p1;
    }
    // same-wave LDS RAW: DS ops are in-order per wave; compiler waits DScnt.

    // ---- y += P * V  (A-frag from LDS, V^T B-frags from staging) -----
    v16bf pa = cat8(*(const v8bf*)&P[lm * 32 + hl * 16],
                    *(const v8bf*)&P[lm * 32 + hl * 16 + 8]);
#pragma unroll
    for (int t = 0; t < 8; ++t) {
      v16bf vb = *(const v16bf*)&Vs[bi][(t * 16 + lm) * 32 + hl * 16];
      yacc[t] = wmma_bf16(pa, vb, yacc[t]);
    }
  }

  // ---- epilogue: normalize, store Y [B,N,CI] bf16 --------------------
  float rinv[8];
#pragma unroll
  for (int v = 0; v < 8; ++v) rinv[v] = 1.f / lrow[v];
#pragma unroll
  for (int t = 0; t < 8; ++t) {
    const int d = t * 16 + lm;
#pragma unroll
    for (int v = 0; v < 8; ++v) {
      int n = qr + v + hl * 8;
      Y[((size_t)b * HW + n) * CI + d] = (__bf16)(yacc[t][v] * rinv[v]);
    }
  }
}

// ---------------------------------------------------------------------------
// Kernel 3: output projection (Y x Wz^T) + BatchNorm2d affine + residual.
// ---------------------------------------------------------------------------
__global__ __launch_bounds__(128) void outproj_kernel(
    const __bf16* __restrict__ Y, const __bf16* __restrict__ Wz,
    const float* __restrict__ g1, const float* __restrict__ g2,
    const float* __restrict__ x, float* __restrict__ out) {
  const int b = blockIdx.y;
  const int tid = threadIdx.x;
  const int lane = tid & 31, wid = tid >> 5;
  const int lm = lane & 15, hl = lane >> 4;
  const int nr = blockIdx.x * 64 + wid * 16;

  const __bf16* Yb = Y + (size_t)b * HW * CI;
  v16bf af[4];
  {
    const size_t rbase = (size_t)(nr + lm) * CI;
#pragma unroll
    for (int kc = 0; kc < 4; ++kc) {
      const v8bf* p0 = (const v8bf*)&Yb[rbase + kc * 32 + hl * 16];
      const v8bf* p1 = (const v8bf*)&Yb[rbase + kc * 32 + hl * 16 + 8];
      af[kc] = cat8(*p0, *p1);
    }
  }

#pragma unroll
  for (int t = 0; t < 16; ++t) {
    const int c = t * 16 + lm;            // output channel
    v8f acc = {};
#pragma unroll
    for (int kc = 0; kc < 4; ++kc) {
      v16bf wb = *(const v16bf*)&Wz[(size_t)c * CI + kc * 32 + hl * 16];
      acc = wmma_bf16(af[kc], wb, acc);
    }
    const float a1 = g1[c], a2 = g2[c];
#pragma unroll
    for (int v = 0; v < 8; ++v) {
      int n = nr + v + hl * 8;
      size_t idx = ((size_t)b * CIN + c) * HW + n;
      out[idx] = acc[v] * a1 + a2 + x[idx];
    }
  }
}

// ---------------------------------------------------------------------------
// Launch
// ---------------------------------------------------------------------------
extern "C" void kernel_launch(void* const* d_in, const int* in_sizes, int n_in,
                              void* d_out, int out_size, void* d_ws, size_t ws_size,
                              hipStream_t stream) {
  const float* x        = (const float*)d_in[0];
  const float* theta_w  = (const float*)d_in[1];
  const float* theta_b  = (const float*)d_in[2];
  const float* t_g      = (const float*)d_in[3];
  const float* t_be     = (const float*)d_in[4];
  const float* t_m      = (const float*)d_in[5];
  const float* t_v      = (const float*)d_in[6];
  const float* phi_w    = (const float*)d_in[7];
  const float* phi_b    = (const float*)d_in[8];
  const float* p_g      = (const float*)d_in[9];
  const float* p_be     = (const float*)d_in[10];
  const float* p_m      = (const float*)d_in[11];
  const float* p_v      = (const float*)d_in[12];
  const float* g_w      = (const float*)d_in[13];
  const float* g_b      = (const float*)d_in[14];
  const float* wz_w     = (const float*)d_in[15];
  const float* wz_b     = (const float*)d_in[16];
  const float* bn_g     = (const float*)d_in[17];
  const float* bn_be    = (const float*)d_in[18];
  const float* bn_m     = (const float*)d_in[19];
  const float* bn_v     = (const float*)d_in[20];
  float* out = (float*)d_out;

  char* ws = (char*)d_ws;
  __bf16* Wq = (__bf16*)(ws + 0);
  __bf16* Wk = (__bf16*)(ws + 65536);
  __bf16* Wv = (__bf16*)(ws + 131072);
  __bf16* Wz = (__bf16*)(ws + 196608);
  float*  bq = (float*)(ws + 262144);
  float*  bk = (float*)(ws + 262656);
  float*  bv = (float*)(ws + 263168);
  float*  g1 = (float*)(ws + 263680);
  float*  g2 = (float*)(ws + 264704);
  __bf16* Qb = (__bf16*)(ws + 265728);
  __bf16* Kb = (__bf16*)(ws + 265728 + 4194304);
  __bf16* Vt = (__bf16*)(ws + 265728 + 2 * 4194304);
  __bf16* Yb = (__bf16*)(ws + 265728 + 3 * 4194304);

  prep_kernel<<<64, 256, 0, stream>>>(
      theta_w, theta_b, t_g, t_be, t_m, t_v,
      phi_w, phi_b, p_g, p_be, p_m, p_v,
      g_w, g_b, wz_w, wz_b, bn_g, bn_be, bn_m, bn_v,
      Wq, Wk, Wv, Wz, bq, bk, bv, g1, g2);

  dim3 grid(HW / 64, NB);
  qkv_kernel<<<grid, 128, 0, stream>>>(x, Wq, Wk, Wv, bq, bk, bv, Qb, Kb, Vt);
  attn_kernel<<<grid, 128, 0, stream>>>(Qb, Kb, Vt, Yb);
  outproj_kernel<<<grid, 128, 0, stream>>>(Yb, Wz, g1, g2, x, out);
}
